// SPModel_12249246728423
// MI455X (gfx1250) — matrix operand order repaired and checked by reading, not compiled, not run on hardware
//
#include <hip/hip_runtime.h>

// ---------------------------------------------------------------------------
// Model constants (match reference)
// ---------------------------------------------------------------------------
static constexpr int Lc = 12;
static constexpr int Dc = 1024;
static constexpr int Hc = 16;
static constexpr int Tc = 1024;
static constexpr int Bc = 2;
static constexpr int Rc = 32;
static constexpr int Mc = Bc * Tc;      // 2048 token rows
static constexpr int HDc = Dc / Hc;     // 64
static constexpr float LORA_SCALE = 2.0f;

// ---------------------------------------------------------------------------
// CDNA5 WMMA types/helpers (wave32, v_wmma_f32_16x16x32_bf16)
// ---------------------------------------------------------------------------
typedef __attribute__((ext_vector_type(8)))  float   v8f;
typedef __attribute__((ext_vector_type(8)))  __bf16  v8bf;
typedef __attribute__((ext_vector_type(16))) __bf16  v16bf;

// Native bf16 conversions: let the backend select the hardware cvt op.
static __device__ inline unsigned pack2bf(float a, float b) {
  union { __bf16 h[2]; unsigned u; } t;
  t.h[0] = (__bf16)a; t.h[1] = (__bf16)b;
  return t.u;
}
static __device__ inline unsigned short bf1(float f) {
  union { __bf16 h; unsigned short s; } t;
  t.h = (__bf16)f;
  return t.s;
}

static __device__ inline v8f vzero8() {
  v8f z;
#pragma unroll
  for (int e = 0; e < 8; ++e) z[e] = 0.0f;
  return z;
}

// Build a 16-half A/B fragment from two contiguous 8-half (16B) LDS chunks.
static __device__ inline v16bf pack16(const unsigned short* p0, const unsigned short* p1) {
  union { v16bf v; v8bf h[2]; } u;
  u.h[0] = *(const v8bf*)p0;
  u.h[1] = *(const v8bf*)p1;
  return u.v;
}

static __device__ inline v8f wmma_bf16(v16bf a, v16bf b, v8f c) {
  // (neg_a, A, neg_b, B, c_mod, C, reuse_a, reuse_b)
  return __builtin_amdgcn_wmma_f32_16x16x32_bf16(false, a, false, b, (short)0, c, false, false);
}

// ---------------------------------------------------------------------------
// Embedding: h[m,d] = wte[ids[m],d] + wpe[m%T,d]
// ---------------------------------------------------------------------------
__global__ __launch_bounds__(256) void embed_kernel(
    const int* __restrict__ ids, const float* __restrict__ wte,
    const float* __restrict__ wpe, float* __restrict__ h) {
  const int m = blockIdx.x;
  const int t = m % Tc;
  const int tok = ids[m];
  const float* src = wte + (size_t)tok * Dc;
  const float* pos = wpe + (size_t)t * Dc;
  float* dst = h + (size_t)m * Dc;
  for (int d = threadIdx.x; d < Dc; d += blockDim.x) dst[d] = src[d] + pos[d];
}

// ---------------------------------------------------------------------------
// LayerNorm over D=1024, one row per block (256 threads, 4 elems/thread)
// ---------------------------------------------------------------------------
__global__ __launch_bounds__(256) void ln_kernel(
    const float* __restrict__ in, const float* __restrict__ w,
    const float* __restrict__ b, float* __restrict__ out) {
  __shared__ float r1[256];
  __shared__ float r2[256];
  const int m = blockIdx.x, tid = threadIdx.x;
  const float* row = in + (size_t)m * Dc;
  float4 d = *(const float4*)(row + tid * 4);
  float s = d.x + d.y + d.z + d.w;
  float q = d.x * d.x + d.y * d.y + d.z * d.z + d.w * d.w;
  r1[tid] = s; r2[tid] = q;
  __syncthreads();
#pragma unroll
  for (int off = 128; off > 0; off >>= 1) {
    if (tid < off) { r1[tid] += r1[tid + off]; r2[tid] += r2[tid + off]; }
    __syncthreads();
  }
  const float mean = r1[0] * (1.0f / Dc);
  const float var  = r2[0] * (1.0f / Dc) - mean * mean;
  const float rstd = rsqrtf(var + 1e-5f);
  float* orow = out + (size_t)m * Dc;
  const int c = tid * 4;
  float4 ww = *(const float4*)(w + c);
  float4 bb = *(const float4*)(b + c);
  float4 o;
  o.x = (d.x - mean) * rstd * ww.x + bb.x;
  o.y = (d.y - mean) * rstd * ww.y + bb.y;
  o.z = (d.z - mean) * rstd * ww.z + bb.z;
  o.w = (d.w - mean) * rstd * ww.w + bb.w;
  *(float4*)(orow + c) = o;
}

// ---------------------------------------------------------------------------
// GEMM: out = act( A[M,K] @ W[N,K]^T + bias + LORA_SCALE * U[M,32] @ LB[N,32]^T
//                  ) [+ res]
// bf16 WMMA 16x16x32, fp32 accumulate. LoRA fused as one extra K-step.
// Double-buffered LDS software pipeline: global loads for step k+1 issue
// before the WMMA block of step k; one barrier per iteration.
// ---------------------------------------------------------------------------
static constexpr int KT  = 32;  // K per stage
static constexpr int LDT = 40;  // padded LDS row stride in halves (80B, 16B-mult)

template <int WAVES_M, int WAVES_N, int WTM, int WTN>
__global__ __launch_bounds__(256) void gemm_kernel(
    const float* __restrict__ A, const float* __restrict__ W,
    const float* __restrict__ bias,
    const float* __restrict__ U, const float* __restrict__ LB,
    const float* __restrict__ res, float* __restrict__ out,
    int M, int N, int K, int act_gelu) {
  constexpr int BM = WAVES_M * WTM * 16;
  constexpr int BN = WAVES_N * WTN * 16;
  constexpr int AG = (BM * 8) / 256;  // float4 granules per thread (A tile)
  constexpr int BG = (BN * 8) / 256;  // float4 granules per thread (B tile)
  __shared__ unsigned short lA[2][BM * LDT];
  __shared__ unsigned short lB[2][BN * LDT];

  const int tid = threadIdx.x;
  const int lane = tid & 31, wid = tid >> 5;
  const int wm = wid / WAVES_N, wn = wid % WAVES_N;
  const int l15 = lane & 15, lh = lane >> 4;
  const int bm0 = blockIdx.y * BM;
  const int bn0 = blockIdx.x * BN;

  v8f acc[WTM][WTN];
#pragma unroll
  for (int i = 0; i < WTM; ++i)
#pragma unroll
    for (int j = 0; j < WTN; ++j) acc[i][j] = vzero8();

  const int nk = K / KT;
  const int total = nk + (U ? 1 : 0);

  float4 ga[AG], gb[BG];

  auto loadG = [&](int kk) {
    const bool lora = (kk >= nk);
    const float* aSrc = lora ? U : A;
    const float* bSrc = lora ? LB : W;
    const int lda = lora ? Rc : K;
    const int ldb = lora ? Rc : K;
    const int k0 = lora ? 0 : kk * KT;
#pragma unroll
    for (int g = 0; g < AG; ++g) {
      const int i = tid + g * 256;
      const int r = i >> 3, c4 = (i & 7) * 4;
      ga[g] = *(const float4*)(aSrc + (size_t)(bm0 + r) * lda + k0 + c4);
    }
#pragma unroll
    for (int g = 0; g < BG; ++g) {
      const int i = tid + g * 256;
      const int r = i >> 3, c4 = (i & 7) * 4;
      gb[g] = *(const float4*)(bSrc + (size_t)(bn0 + r) * ldb + k0 + c4);
    }
  };

  auto storeL = [&](int buf, float asc) {
#pragma unroll
    for (int g = 0; g < AG; ++g) {
      const int i = tid + g * 256;
      const int r = i >> 3, c4 = (i & 7) * 4;
      uint2 p;
      p.x = pack2bf(ga[g].x * asc, ga[g].y * asc);
      p.y = pack2bf(ga[g].z * asc, ga[g].w * asc);
      *(uint2*)&lA[buf][r * LDT + c4] = p;      // single ds_store_b64
    }
#pragma unroll
    for (int g = 0; g < BG; ++g) {
      const int i = tid + g * 256;
      const int r = i >> 3, c4 = (i & 7) * 4;
      uint2 p;
      p.x = pack2bf(gb[g].x, gb[g].y);
      p.y = pack2bf(gb[g].z, gb[g].w);
      *(uint2*)&lB[buf][r * LDT + c4] = p;
    }
  };

  loadG(0);
  for (int kk = 0; kk < total; ++kk) {
    const int buf = kk & 1;
    storeL(buf, (kk >= nk) ? LORA_SCALE : 1.0f);
    __syncthreads();
    if (kk + 1 < total) loadG(kk + 1);            // hide HBM latency behind WMMA
    if (kk + 2 < nk) {                            // gfx1250 global_prefetch_b8
      __builtin_prefetch(W + (size_t)(bn0 + (tid % BN)) * K + (size_t)(kk + 2) * KT, 0, 0);
    }

    // Fragment loads per ISA 16-bit A (16x32) / B (32x16) layouts
    v16bf af[WTM], bfr[WTN];
#pragma unroll
    for (int i = 0; i < WTM; ++i) {
      const int row = wm * WTM * 16 + i * 16 + l15;
      const int kb = lh * 8;
      af[i] = pack16(&lA[buf][row * LDT + kb], &lA[buf][row * LDT + kb + 16]);
    }
#pragma unroll
    for (int j = 0; j < WTN; ++j) {
      const int col = wn * WTN * 16 + j * 16 + l15;
      const int kb = lh * 8;
      bfr[j] = pack16(&lB[buf][col * LDT + kb], &lB[buf][col * LDT + kb + 16]);
    }
#pragma unroll
    for (int i = 0; i < WTM; ++i)
#pragma unroll
      for (int j = 0; j < WTN; ++j)
        acc[i][j] = wmma_bf16(af[i], bfr[j], acc[i][j]);
    // No trailing barrier: next iteration writes the other buffer; the
    // syncthreads (s_wait_dscnt 0) above orders reads vs. two-step-later writes.
  }

  // Epilogue: bias, GELU(exact), optional residual
#pragma unroll
  for (int i = 0; i < WTM; ++i) {
#pragma unroll
    for (int j = 0; j < WTN; ++j) {
      const int colT = bn0 + wn * WTN * 16 + j * 16 + l15;
      const float bia = bias ? bias[colT] : 0.0f;
#pragma unroll
      for (int r = 0; r < 8; ++r) {
        const int rowT = bm0 + wm * WTM * 16 + i * 16 + r + (lh << 3);
        float v = acc[i][j][r] + bia;
        if (act_gelu) v = 0.5f * v * (1.0f + erff(v * 0.70710678118654752f));
        const size_t idx = (size_t)rowT * N + colT;
        if (res) v += res[idx];
        out[idx] = v;
      }
    }
  }
}

// ---------------------------------------------------------------------------
// Flash attention: one block = (b, h, 128 q rows); 8 waves x 16 q rows each.
// Online softmax; WMMA for S=Q·K^T and O=P·V via LDS-staged bf16 tiles.
// ---------------------------------------------------------------------------
__global__ __launch_bounds__(256) void attn_kernel(
    const float* __restrict__ qkv, float* __restrict__ att) {
  __shared__ unsigned short lK[32 * 72];      // K tile  [kk][d], stride 72
  __shared__ unsigned short lVT[64 * 40];     // V^T     [d][kk], stride 40
  __shared__ unsigned short lP[8 * 16 * 40];  // per-wave P [16][32], stride 40

  const int tid = threadIdx.x, lane = tid & 31, wid = tid >> 5;
  const int l15 = lane & 15, lh = lane >> 4;
  const int bh = blockIdx.y;
  const int b = bh / Hc, hh = bh % Hc;
  const int q0 = blockIdx.x * 128;
  const int qrow0 = q0 + wid * 16;

  const size_t ldq = (size_t)(3 * Dc);
  const float* qbase = qkv + (size_t)b * Tc * 3 * Dc + hh * HDc;
  const float* kbase = qbase + Dc;
  const float* vbase = qbase + 2 * Dc;

  // Q fragments (16 rows x 64 d, pre-scaled by 1/sqrt(HD)=0.125)
  v16bf qf[2];
  {
    const float* qr = qbase + (size_t)(qrow0 + l15) * ldq;
#pragma unroll
    for (int s2 = 0; s2 < 2; ++s2) {
      const int c1 = s2 * 32 + lh * 8;
      const int c2 = c1 + 16;
      float4 fa = *(const float4*)(qr + c1);
      float4 fb = *(const float4*)(qr + c1 + 4);
      float4 fc = *(const float4*)(qr + c2);
      float4 fd = *(const float4*)(qr + c2 + 4);
      union { v16bf v; unsigned u[8]; } t;
      t.u[0] = pack2bf(fa.x * 0.125f, fa.y * 0.125f);
      t.u[1] = pack2bf(fa.z * 0.125f, fa.w * 0.125f);
      t.u[2] = pack2bf(fb.x * 0.125f, fb.y * 0.125f);
      t.u[3] = pack2bf(fb.z * 0.125f, fb.w * 0.125f);
      t.u[4] = pack2bf(fc.x * 0.125f, fc.y * 0.125f);
      t.u[5] = pack2bf(fc.z * 0.125f, fc.w * 0.125f);
      t.u[6] = pack2bf(fd.x * 0.125f, fd.y * 0.125f);
      t.u[7] = pack2bf(fd.z * 0.125f, fd.w * 0.125f);
      qf[s2] = t.v;
    }
  }

  float mrow[8], lrow[8];
  v8f oacc[4];
#pragma unroll
  for (int r = 0; r < 8; ++r) { mrow[r] = -1e30f; lrow[r] = 0.0f; }
#pragma unroll
  for (int t2 = 0; t2 < 4; ++t2) oacc[t2] = vzero8();

  unsigned short* lPw = lP + wid * 16 * 40;
  const int steps = (q0 + 128) / 32;

  for (int s = 0; s < steps; ++s) {
    const int kb = s * 32;
    // Stage K (row-major, packed b64 stores) and V (transposed) tiles
#pragma unroll
    for (int t2 = 0; t2 < 2; ++t2) {
      const int i = tid * 2 + t2;            // 0..511 float4 granules
      const int kkr = i >> 4, c4 = (i & 15) * 4;
      float4 fk = *(const float4*)(kbase + (size_t)(kb + kkr) * ldq + c4);
      uint2 pk;
      pk.x = pack2bf(fk.x, fk.y);
      pk.y = pack2bf(fk.z, fk.w);
      *(uint2*)&lK[kkr * 72 + c4] = pk;
      float4 fv = *(const float4*)(vbase + (size_t)(kb + kkr) * ldq + c4);
      lVT[(c4 + 0) * 40 + kkr] = bf1(fv.x);
      lVT[(c4 + 1) * 40 + kkr] = bf1(fv.y);
      lVT[(c4 + 2) * 40 + kkr] = bf1(fv.z);
      lVT[(c4 + 3) * 40 + kkr] = bf1(fv.w);
    }
    __syncthreads();

    // S = Q K^T  (two 16-col tiles, K-dim 64 = 2 wmma each)
    v8f sacc[2];
    sacc[0] = vzero8(); sacc[1] = vzero8();
#pragma unroll
    for (int j = 0; j < 2; ++j) {
      const int colr = 16 * j + l15;
#pragma unroll
      for (int ds2 = 0; ds2 < 2; ++ds2) {
        const int db = lh * 8 + 32 * ds2;
        v16bf kf = pack16(&lK[colr * 72 + db], &lK[colr * 72 + db + 16]);
        sacc[j] = wmma_bf16(qf[ds2], kf, sacc[j]);
      }
    }

    // Online softmax update; write P (bf16) to per-wave LDS
#pragma unroll
    for (int r = 0; r < 8; ++r) {
      const int grow = qrow0 + r + (lh << 3);
      float s0 = (kb + l15 > grow)      ? -1e30f : sacc[0][r];
      float s1 = (kb + 16 + l15 > grow) ? -1e30f : sacc[1][r];
      float rm = fmaxf(s0, s1);
#pragma unroll
      for (int m2 = 1; m2 < 16; m2 <<= 1) rm = fmaxf(rm, __shfl_xor(rm, m2, 32));
      const float mn = fmaxf(mrow[r], rm);
      const float sc_old = __expf(mrow[r] - mn);
      const float p0 = __expf(s0 - mn), p1 = __expf(s1 - mn);
      float rs = p0 + p1;
#pragma unroll
      for (int m2 = 1; m2 < 16; m2 <<= 1) rs += __shfl_xor(rs, m2, 32);
      lrow[r] = lrow[r] * sc_old + rs;
      mrow[r] = mn;
#pragma unroll
      for (int t2 = 0; t2 < 4; ++t2) oacc[t2][r] *= sc_old;
      const int rl = r + (lh << 3);
      lPw[rl * 40 + l15] = bf1(p0);
      lPw[rl * 40 + 16 + l15] = bf1(p1);
    }
    // Per-wave LDS RAW: make P stores visible before fragment reload
    asm volatile("s_wait_dscnt 0" ::: "memory");

    // O += P · V
    v16bf pf;
    {
      const int kb2 = lh * 8;
      pf = pack16(&lPw[l15 * 40 + kb2], &lPw[l15 * 40 + kb2 + 16]);
    }
#pragma unroll
    for (int t2 = 0; t2 < 4; ++t2) {
      const int cold = 16 * t2 + l15;
      const int kkb = lh * 8;
      v16bf vf = pack16(&lVT[cold * 40 + kkb], &lVT[cold * 40 + kkb + 16]);
      oacc[t2] = wmma_bf16(pf, vf, oacc[t2]);
    }
    __syncthreads();
  }

  // Normalize and write O to att[M, D] at this head's columns
#pragma unroll
  for (int t2 = 0; t2 < 4; ++t2) {
    const int col = hh * HDc + 16 * t2 + l15;
#pragma unroll
    for (int r = 0; r < 8; ++r) {
      const int row = qrow0 + r + (lh << 3);
      att[((size_t)(b * Tc + row)) * Dc + col] = oacc[t2][r] / lrow[r];
    }
  }
}

// ---------------------------------------------------------------------------
// Host-side orchestration (graph-capture safe: only kernel launches on stream)
// ---------------------------------------------------------------------------
extern "C" void kernel_launch(void* const* d_in, const int* in_sizes, int n_in,
                              void* d_out, int out_size, void* d_ws, size_t ws_size,
                              hipStream_t stream) {
  (void)in_sizes; (void)n_in; (void)out_size; (void)ws_size;
  const int*   ids    = (const int*)d_in[0];
  const float* wte    = (const float*)d_in[1];
  const float* wpe    = (const float*)d_in[2];
  const float* ln1_w  = (const float*)d_in[3];
  const float* ln1_b  = (const float*)d_in[4];
  const float* attn_w = (const float*)d_in[5];
  const float* attn_b = (const float*)d_in[6];
  const float* attn_la= (const float*)d_in[7];
  const float* attn_lb= (const float*)d_in[8];
  const float* proj_w = (const float*)d_in[9];
  const float* proj_b = (const float*)d_in[10];
  const float* proj_la= (const float*)d_in[11];
  const float* proj_lb= (const float*)d_in[12];
  const float* ln2_w  = (const float*)d_in[13];
  const float* ln2_b  = (const float*)d_in[14];
  const float* fc_w   = (const float*)d_in[15];
  const float* fc_b   = (const float*)d_in[16];
  const float* fc_la  = (const float*)d_in[17];
  const float* fc_lb  = (const float*)d_in[18];
  const float* mp_w   = (const float*)d_in[19];
  const float* mp_b   = (const float*)d_in[20];
  const float* mp_la  = (const float*)d_in[21];
  const float* mp_lb  = (const float*)d_in[22];
  const float* lnf_w  = (const float*)d_in[23];
  const float* lnf_b  = (const float*)d_in[24];

  // Workspace layout (fp32): h, x, qkv, att, mbuf, u
  float* h    = (float*)d_ws;
  float* x    = h    + (size_t)Mc * Dc;
  float* qkv  = x    + (size_t)Mc * Dc;
  float* att  = qkv  + (size_t)Mc * 3 * Dc;
  float* mbuf = att  + (size_t)Mc * Dc;
  float* u    = mbuf + (size_t)Mc * 4 * Dc;

  const dim3 blk(256);
  // Main GEMM: 128x256 block tile (2x4 waves, 4x4 tiles each)
  auto gemm_main = gemm_kernel<2, 4, 4, 4>;
  // LoRA-rank GEMM: 256x32 block tile (8x1 waves, 2x2 tiles each)
  auto gemm_rank = gemm_kernel<8, 1, 2, 2>;

  embed_kernel<<<Mc, blk, 0, stream>>>(ids, wte, wpe, h);

  for (int l = 0; l < Lc; ++l) {
    const size_t lD  = (size_t)l * Dc;
    // x = LN(h, ln1)
    ln_kernel<<<Mc, blk, 0, stream>>>(h, ln1_w + lD, ln1_b + lD, x);
    // u = x @ attn_la^T   (N=32, K=D)
    gemm_rank<<<dim3(1, Mc / 256), blk, 0, stream>>>(
        x, attn_la + (size_t)l * Rc * Dc, nullptr, nullptr, nullptr, nullptr,
        u, Mc, Rc, Dc, 0);
    // qkv = x @ attn_w^T + attn_b + 2*(u @ attn_lb^T)
    gemm_main<<<dim3(3 * Dc / 256, Mc / 128), blk, 0, stream>>>(
        x, attn_w + (size_t)l * 3 * Dc * Dc, attn_b + (size_t)l * 3 * Dc,
        u, attn_lb + (size_t)l * 3 * Dc * Rc, nullptr, qkv, Mc, 3 * Dc, Dc, 0);
    // att = causal softmax attention
    attn_kernel<<<dim3(Tc / 128, Bc * Hc), blk, 0, stream>>>(qkv, att);
    // u = att @ proj_la^T
    gemm_rank<<<dim3(1, Mc / 256), blk, 0, stream>>>(
        att, proj_la + (size_t)l * Rc * Dc, nullptr, nullptr, nullptr, nullptr,
        u, Mc, Rc, Dc, 0);
    // h = h + att @ proj_w^T + proj_b + 2*(u @ proj_lb^T)
    gemm_main<<<dim3(Dc / 256, Mc / 128), blk, 0, stream>>>(
        att, proj_w + (size_t)l * Dc * Dc, proj_b + lD,
        u, proj_lb + (size_t)l * Dc * Rc, h, h, Mc, Dc, Dc, 0);
    // x = LN(h, ln2)
    ln_kernel<<<Mc, blk, 0, stream>>>(h, ln2_w + lD, ln2_b + lD, x);
    // u = x @ fc_la^T
    gemm_rank<<<dim3(1, Mc / 256), blk, 0, stream>>>(
        x, fc_la + (size_t)l * Rc * Dc, nullptr, nullptr, nullptr, nullptr,
        u, Mc, Rc, Dc, 0);
    // mbuf = gelu(x @ fc_w^T + fc_b + 2*(u @ fc_lb^T))
    gemm_main<<<dim3(4 * Dc / 256, Mc / 128), blk, 0, stream>>>(
        x, fc_w + (size_t)l * 4 * Dc * Dc, fc_b + (size_t)l * 4 * Dc,
        u, fc_lb + (size_t)l * 4 * Dc * Rc, nullptr, mbuf, Mc, 4 * Dc, Dc, 1);
    // u = mbuf @ mp_la^T  (K=4D)
    gemm_rank<<<dim3(1, Mc / 256), blk, 0, stream>>>(
        mbuf, mp_la + (size_t)l * Rc * 4 * Dc, nullptr, nullptr, nullptr, nullptr,
        u, Mc, Rc, 4 * Dc, 0);
    // h = h + mbuf @ mp_w^T + mp_b + 2*(u @ mp_lb^T)
    gemm_main<<<dim3(Dc / 256, Mc / 128), blk, 0, stream>>>(
        mbuf, mp_w + (size_t)l * Dc * 4 * Dc, mp_b + lD,
        u, mp_lb + (size_t)l * Dc * Rc, h, h, Mc, Dc, 4 * Dc, 0);
  }

  // out = LN(h, lnf)
  ln_kernel<<<Mc, blk, 0, stream>>>(h, lnf_w, lnf_b, (float*)d_out);
}
